// Indexer_47802986004847
// MI455X (gfx1250) — matrix operand order repaired and checked by reading, not compiled, not run on hardware
//
#include <hip/hip_runtime.h>
#include <hip/hip_bf16.h>

#define T_ 4096
#define HIDDEN_ 7168
#define QLORA_ 1536
#define NHEAD_ 64
#define HDIM_ 128
#define TOPK_ 2048
#define NEG_INF_ (-1e30f)

typedef __attribute__((ext_vector_type(16))) __bf16 v16bf;
typedef __attribute__((ext_vector_type(8)))  float  v8f;
typedef __attribute__((ext_vector_type(4)))  unsigned int v4u;
typedef __attribute__((ext_vector_type(8)))  int v8i_;
typedef __attribute__((ext_vector_type(4)))  int v4i_;

union V16U { v16bf v; uint4 q[2]; unsigned short s[16]; };

static __device__ inline unsigned short f2bf(float f) {
  __hip_bfloat16 h = __float2bfloat16(f);
  return *reinterpret_cast<unsigned short*>(&h);
}

static __device__ inline v8f wmma_bf16(v16bf a, v16bf b, v8f c) {
  return __builtin_amdgcn_wmma_f32_16x16x32_bf16(
      false, a, false, b, (short)0, c, false, false);
}

// A operand: 16x32 bf16 from row-major storage (row stride ld elems).
static __device__ inline v16bf load_A_rm(const unsigned short* p, int ld) {
  int lane = threadIdx.x & 31;
  int m = lane & 15, half = lane >> 4;
  const unsigned short* row = p + m * ld;
  V16U u;
  u.q[0] = *(const uint4*)(row + 8 * half);
  u.q[1] = *(const uint4*)(row + 16 + 8 * half);
  return u.v;
}

// B operand (32x16): B[k][n] = src[n][k], src is N-major rows of length ld.
static __device__ inline v16bf load_Bt_rm(const unsigned short* src, int ld, int k0) {
  int lane = threadIdx.x & 31;
  int n = lane & 15, half = lane >> 4;
  const unsigned short* row = src + (size_t)n * ld + k0 + 8 * half;
  V16U u;
  u.q[0] = *(const uint4*)(row);
  u.q[1] = *(const uint4*)(row + 16);
  return u.v;
}

// TDM: DMA a 64-row x 256B bf16 tile (row stride 8192 elems) from global to LDS.
// D# per CDNA5 ISA 8.3/8.4: group0 {count=1, lds_addr, global_addr[56:0], type=2},
// group1 {data_size=2B, tensor_dim0=128, tensor_dim1=64, tile_dim0=128, tile_dim1=64,
//         tensor_dim0_stride=8192}.
static __device__ inline void tdm_load_q(unsigned lds_off, const unsigned short* gptr) {
#if __has_builtin(__builtin_amdgcn_tensor_load_to_lds)
  unsigned long long ga = (unsigned long long)(size_t)gptr;
  v4u g0 = {1u, lds_off, (unsigned)ga,
            (unsigned)((ga >> 32) & 0x01FFFFFFu) | (2u << 30)};
  v8i_ g1 = {0x10000,               // data_size = 1 (2 bytes)
             (int)(128u << 16),     // tensor_dim0[15:0]
             (int)(64u << 16),      // tensor_dim1[15:0]
             (int)(128u << 16),     // tile_dim0
             64,                    // tile_dim1
             8192,                  // tensor_dim0_stride[31:0]
             0, 0};
  v4i_ gz = {0, 0, 0, 0};
#if defined(__clang_major__) && __clang_major__ >= 23
  v8i_ gz8 = {0, 0, 0, 0, 0, 0, 0, 0};
  __builtin_amdgcn_tensor_load_to_lds(g0, g1, gz, gz, gz8, 0);
#else
  __builtin_amdgcn_tensor_load_to_lds(g0, g1, gz, gz, 0);
#endif
#endif
}

// -------- transpose + f32->bf16: in (K x N) -> out (N x K) --------
__global__ __launch_bounds__(256) void transpose_cvt_kernel(
    const float* __restrict__ in, unsigned short* __restrict__ out, int K, int N) {
  __shared__ float t[32][33];
  const int n0 = blockIdx.x * 32, k0 = blockIdx.y * 32;
  const int tid = threadIdx.x;
  for (int e = tid; e < 1024; e += 256) {
    int r = e >> 5, c = e & 31;
    t[r][c] = in[(size_t)(k0 + r) * N + n0 + c];
  }
  __syncthreads();
  for (int e = tid; e < 1024; e += 256) {
    int r = e >> 5, c = e & 31;
    out[(size_t)(n0 + r) * K + k0 + c] = f2bf(t[c][r]);
  }
}

// stage nrows x 128 f32 tile as bf16 into LDS; seg = 8 elems
static __device__ inline void stage_act_bf16(unsigned short* dst, const float* src,
                                             int ld_src, int nsegs, int nthreads) {
  for (int e = threadIdx.x; e < nsegs; e += nthreads) {
    int row = e >> 4, seg = e & 15;
    const float* g = src + (size_t)row * ld_src + seg * 8;
    unsigned short* d = dst + row * 128 + seg * 8;
#pragma unroll
    for (int q = 0; q < 8; ++q) d[q] = f2bf(g[q]);
  }
}

// ---------------- k = RoPE(LN(hidden @ Wk)) -> bf16 (T,128) ----------------
__global__ __launch_bounds__(256) void k_proj_kernel(
    const float* __restrict__ hidden, const unsigned short* __restrict__ WkT,
    const float* __restrict__ gamma, const float* __restrict__ beta,
    unsigned short* __restrict__ k_full) {
  __shared__ __align__(16) unsigned short stageA[16 * 128];
  __shared__ __align__(16) float tile[16 * 128];
  __shared__ float redS[16 * 16], redS2[16 * 16];
  __shared__ float muS[16], rsS[16];
  const int i0 = blockIdx.x * 16;
  const int tid = threadIdx.x;
  const int wave = tid >> 5;
  v8f acc = {};
  for (int cc = 0; cc < HIDDEN_ / 128; ++cc) {
    __syncthreads();
    stage_act_bf16(stageA, hidden + (size_t)i0 * HIDDEN_ + cc * 128, HIDDEN_, 256, 256);
    __syncthreads();
#pragma unroll
    for (int s = 0; s < 4; ++s) {
      v16bf a = load_A_rm(stageA + s * 32, 128);
      v16bf b = load_Bt_rm(WkT + (size_t)(wave * 16) * HIDDEN_, HIDDEN_, cc * 128 + s * 32);
      acc = wmma_bf16(a, b, acc);
    }
  }
  {
    int lane = tid & 31, n = lane & 15, half = lane >> 4;
#pragma unroll
    for (int r = 0; r < 8; ++r)
      tile[(r + 8 * half) * 128 + wave * 16 + n] = acc[r];
  }
  __syncthreads();
  {
    int row = tid >> 4, g = tid & 15;
    float s = 0.f, s2 = 0.f;
#pragma unroll
    for (int c = 0; c < 8; ++c) {
      float x = tile[row * 128 + g * 8 + c];
      s += x; s2 += x * x;
    }
    redS[row * 16 + g] = s; redS2[row * 16 + g] = s2;
  }
  __syncthreads();
  if (tid < 16) {
    float s = 0.f, s2 = 0.f;
    for (int g = 0; g < 16; ++g) { s += redS[tid * 16 + g]; s2 += redS2[tid * 16 + g]; }
    float mu = s * (1.f / 128.f);
    float var = s2 * (1.f / 128.f) - mu * mu;
    muS[tid] = mu; rsS[tid] = rsqrtf(var + 1e-6f);
  }
  __syncthreads();
  {
    int row = tid >> 4, g = tid & 15;
    float mu = muS[row], rs = rsS[row];
#pragma unroll
    for (int c = 0; c < 8; ++c) {
      int col = g * 8 + c;
      float x = tile[row * 128 + col];
      tile[row * 128 + col] = (x - mu) * rs * gamma[col] + beta[col];
    }
  }
  __syncthreads();
  for (int t = tid; t < 16 * 32; t += 256) {
    int row = t >> 5, c = t & 31;
    float ang = (float)(i0 + row) * powf(10000.f, -(float)c / 32.f);
    float sn, cs; sincosf(ang, &sn, &cs);
    float x1 = tile[row * 128 + 2 * c], x2 = tile[row * 128 + 2 * c + 1];
    tile[row * 128 + 2 * c] = x1 * cs - x2 * sn;
    tile[row * 128 + 2 * c + 1] = x1 * sn + x2 * cs;
  }
  __syncthreads();
  for (int t = tid; t < 16 * 128; t += 256)
    k_full[(size_t)(i0 + (t >> 7)) * 128 + (t & 127)] = f2bf(tile[t]);
}

// ------------- q = RoPE_head(qr @ Wq_b) -> bf16 (T, 64*128) -------------
__global__ __launch_bounds__(256) void q_proj_kernel(
    const float* __restrict__ qr, const unsigned short* __restrict__ WqT,
    unsigned short* __restrict__ q_full) {
  __shared__ __align__(16) unsigned short stageA[16 * 128];
  __shared__ __align__(16) float tile[16 * 128];
  const int i0 = blockIdx.x * 16;
  const int h = blockIdx.y;
  const int tid = threadIdx.x;
  const int wave = tid >> 5;
  v8f acc = {};
  for (int cc = 0; cc < QLORA_ / 128; ++cc) {
    __syncthreads();
    stage_act_bf16(stageA, qr + (size_t)i0 * QLORA_ + cc * 128, QLORA_, 256, 256);
    __syncthreads();
#pragma unroll
    for (int s = 0; s < 4; ++s) {
      v16bf a = load_A_rm(stageA + s * 32, 128);
      v16bf b = load_Bt_rm(WqT + (size_t)(h * HDIM_ + wave * 16) * QLORA_, QLORA_,
                           cc * 128 + s * 32);
      acc = wmma_bf16(a, b, acc);
    }
  }
  {
    int lane = tid & 31, n = lane & 15, half = lane >> 4;
#pragma unroll
    for (int r = 0; r < 8; ++r)
      tile[(r + 8 * half) * 128 + wave * 16 + n] = acc[r];
  }
  __syncthreads();
  for (int t = tid; t < 16 * 32; t += 256) {
    int row = t >> 5, c = t & 31;
    float ang = (float)(i0 + row) * powf(10000.f, -(float)c / 32.f);
    float sn, cs; sincosf(ang, &sn, &cs);
    float x1 = tile[row * 128 + 2 * c], x2 = tile[row * 128 + 2 * c + 1];
    tile[row * 128 + 2 * c] = x1 * cs - x2 * sn;
    tile[row * 128 + 2 * c + 1] = x1 * sn + x2 * cs;
  }
  __syncthreads();
  for (int t = tid; t < 16 * 128; t += 256)
    q_full[(size_t)(i0 + (t >> 7)) * (NHEAD_ * HDIM_) + h * HDIM_ + (t & 127)] =
        f2bf(tile[t]);
}

// ---------------- w = hidden @ Wproj * scale -> f32 (T,64) ----------------
__global__ __launch_bounds__(128) void w_proj_kernel(
    const float* __restrict__ hidden, const unsigned short* __restrict__ WpT,
    float* __restrict__ w_out) {
  __shared__ __align__(16) unsigned short stageA[16 * 128];
  const int i0 = blockIdx.x * 16;
  const int tid = threadIdx.x;
  const int wave = tid >> 5;
  v8f acc = {};
  for (int cc = 0; cc < HIDDEN_ / 128; ++cc) {
    __syncthreads();
    stage_act_bf16(stageA, hidden + (size_t)i0 * HIDDEN_ + cc * 128, HIDDEN_, 256, 128);
    __syncthreads();
#pragma unroll
    for (int s = 0; s < 4; ++s) {
      v16bf a = load_A_rm(stageA + s * 32, 128);
      v16bf b = load_Bt_rm(WpT + (size_t)(wave * 16) * HIDDEN_, HIDDEN_, cc * 128 + s * 32);
      acc = wmma_bf16(a, b, acc);
    }
  }
  const float scale = 0.011048543456039806f;  // 128^-0.5 * 64^-0.5
  int lane = tid & 31, n = lane & 15, half = lane >> 4;
#pragma unroll
  for (int r = 0; r < 8; ++r)
    w_out[(size_t)(i0 + r + 8 * half) * NHEAD_ + wave * 16 + n] = acc[r] * scale;
}

// ------- scores[i,j] = sum_h w[i,h]*relu(q_h[i].k[j]) ; causal mask -------
// 64x64 block tile, 8 waves = 4(i) x 2(j), each wave 16x32.
// Per-head q tile (16KB) streamed by TDM, double-buffered against WMMA.
__global__ __launch_bounds__(256) void scores_kernel(
    const unsigned short* __restrict__ q_full,
    const unsigned short* __restrict__ k_full,
    const float* __restrict__ w, float* __restrict__ scores) {
  __shared__ __align__(16) unsigned short q_s[2][64 * 128];  // 2 x 16KB
  __shared__ __align__(16) float w_s[64 * 64];               // 16KB
  const int j0 = blockIdx.x * 64;
  const int i0 = blockIdx.y * 64;
  const int tid = threadIdx.x;

  if (j0 >= i0 + 64) {  // fully above the diagonal: all masked
    for (int t = tid; t < 64 * 64; t += 256)
      scores[(size_t)(i0 + (t >> 6)) * T_ + j0 + (t & 63)] = NEG_INF_;
    return;
  }

  const int wave = tid >> 5;
  const int wi = wave >> 1, wj = wave & 1;
  const int lane = tid & 31, n = lane & 15, half = lane >> 4;

  // k-tile B operands: registers for the whole head loop
  v16bf B[2][4];
#pragma unroll
  for (int jt = 0; jt < 2; ++jt)
#pragma unroll
    for (int kk = 0; kk < 4; ++kk)
      B[jt][kk] = load_Bt_rm(k_full + (size_t)(j0 + 32 * wj + 16 * jt) * 128, 128, kk * 32);

  {  // async-stage w tile: 16KB fully contiguous; thread t copies 64B
    unsigned wloff = (unsigned)(size_t)(&w_s[0]) + (unsigned)(tid * 64);
    unsigned long long ga =
        (unsigned long long)(size_t)(w + (size_t)i0 * NHEAD_ + tid * 16);
    asm volatile(
        "global_load_async_to_lds_b128 %0, %1, off\n\t"
        "global_load_async_to_lds_b128 %0, %1, off offset:16\n\t"
        "global_load_async_to_lds_b128 %0, %1, off offset:32\n\t"
        "global_load_async_to_lds_b128 %0, %1, off offset:48"
        :: "v"(wloff), "v"(ga) : "memory");
    asm volatile("s_wait_asynccnt 0x0" ::: "memory");
  }

  const unsigned q_base = (unsigned)(size_t)(&q_s[0][0]);
  const unsigned short* qg = q_full + (size_t)i0 * (NHEAD_ * HDIM_);
  if (wave == 0) tdm_load_q(q_base, qg);  // prefetch head 0

  v8f acc0 = {}, acc1 = {};
  for (int h = 0; h < NHEAD_; ++h) {
    __syncthreads();  // compute on buf[(h+1)&1] from iteration h-1 is finished
    if (h + 1 < NHEAD_) {
      if (wave == 0)
        tdm_load_q(q_base + (unsigned)(((h + 1) & 1) * (64 * 128 * 2)),
                   qg + (h + 1) * HDIM_);
      __builtin_amdgcn_s_wait_tensorcnt(1);  // tile h landed (TDM in-order per wave)
    } else {
      __builtin_amdgcn_s_wait_tensorcnt(0);
    }
    __syncthreads();  // publish buf[h&1]
    const unsigned short* qb = q_s[h & 1];
    v8f S0 = {}, S1 = {};
#pragma unroll
    for (int kk = 0; kk < 4; ++kk) {
      v16bf a = load_A_rm(qb + (16 * wi) * 128 + kk * 32, 128);
      S0 = wmma_bf16(a, B[0][kk], S0);
      S1 = wmma_bf16(a, B[1][kk], S1);
    }
#pragma unroll
    for (int r = 0; r < 8; ++r) {
      float wv = w_s[(16 * wi + r + 8 * half) * 64 + h];
      acc0[r] += wv * fmaxf(S0[r], 0.f);
      acc1[r] += wv * fmaxf(S1[r], 0.f);
    }
  }
#pragma unroll
  for (int r = 0; r < 8; ++r) {
    int row = i0 + 16 * wi + r + 8 * half;
    int c0 = j0 + 32 * wj + n;
    int c1 = c0 + 16;
    scores[(size_t)row * T_ + c0] = (c0 <= row) ? acc0[r] : NEG_INF_;
    scores[(size_t)row * T_ + c1] = (c1 <= row) ? acc1[r] : NEG_INF_;
  }
}

// ---------- per-row bitonic top-k (desc by val, asc index tiebreak) ----------
__global__ __launch_bounds__(512) void topk_kernel(
    const float* __restrict__ scores, float* __restrict__ out_vals,
    int* __restrict__ out_idx) {
  __shared__ float sv[T_];
  __shared__ int si[T_];
  const int row = blockIdx.x;
  const int tid = threadIdx.x;
  for (int t = tid; t < T_; t += 512) {
    sv[t] = scores[(size_t)row * T_ + t];
    si[t] = t;
  }
  __syncthreads();
  for (int k = 2; k <= T_; k <<= 1) {
    for (int j = k >> 1; j > 0; j >>= 1) {
      for (int t = tid; t < T_ / 2; t += 512) {
        int i = ((t & ~(j - 1)) << 1) | (t & (j - 1));
        int l = i | j;
        float vi = sv[i], vl = sv[l];
        int ii = si[i], il = si[l];
        bool gt = (vi > vl) || (vi == vl && ii < il);
        bool desc = ((i & k) == 0);
        if (gt != desc) { sv[i] = vl; sv[l] = vi; si[i] = il; si[l] = ii; }
      }
      __syncthreads();
    }
  }
  for (int t = tid; t < TOPK_; t += 512) {
    out_vals[(size_t)row * TOPK_ + t] = sv[t];
    out_idx[(size_t)row * TOPK_ + t] = si[t];
  }
}

extern "C" void kernel_launch(void* const* d_in, const int* in_sizes, int n_in,
                              void* d_out, int out_size, void* d_ws, size_t ws_size,
                              hipStream_t stream) {
  const float* hidden = (const float*)d_in[0];
  const float* qr     = (const float*)d_in[1];
  // d_in[2] = positions (arange(T)) — implicit via row indices
  const float* Wq_b   = (const float*)d_in[3];
  const float* Wk     = (const float*)d_in[4];
  const float* Wproj  = (const float*)d_in[5];
  const float* gamma  = (const float*)d_in[6];
  const float* beta   = (const float*)d_in[7];

  char* ws = (char*)d_ws;
  size_t off = 0;
  unsigned short* q_full = (unsigned short*)(ws + off); off += (size_t)T_ * NHEAD_ * HDIM_ * 2;
  unsigned short* k_full = (unsigned short*)(ws + off); off += (size_t)T_ * HDIM_ * 2;
  float* w_ws            = (float*)(ws + off);          off += (size_t)T_ * NHEAD_ * 4;
  float* scores          = (float*)(ws + off);          off += (size_t)T_ * T_ * 4;
  unsigned short* WkT    = (unsigned short*)(ws + off); off += (size_t)HDIM_ * HIDDEN_ * 2;
  unsigned short* WqT    = (unsigned short*)(ws + off); off += (size_t)NHEAD_ * HDIM_ * QLORA_ * 2;
  unsigned short* WpT    = (unsigned short*)(ws + off); off += (size_t)NHEAD_ * HIDDEN_ * 2;

  transpose_cvt_kernel<<<dim3(HDIM_ / 32, HIDDEN_ / 32), 256, 0, stream>>>(
      Wk, WkT, HIDDEN_, HDIM_);
  transpose_cvt_kernel<<<dim3((NHEAD_ * HDIM_) / 32, QLORA_ / 32), 256, 0, stream>>>(
      Wq_b, WqT, QLORA_, NHEAD_ * HDIM_);
  transpose_cvt_kernel<<<dim3(NHEAD_ / 32, HIDDEN_ / 32), 256, 0, stream>>>(
      Wproj, WpT, HIDDEN_, NHEAD_);

  k_proj_kernel<<<T_ / 16, 256, 0, stream>>>(hidden, WkT, gamma, beta, k_full);
  q_proj_kernel<<<dim3(T_ / 16, NHEAD_), 256, 0, stream>>>(qr, WqT, q_full);
  w_proj_kernel<<<T_ / 16, 128, 0, stream>>>(hidden, WpT, w_ws);
  scores_kernel<<<dim3(T_ / 64, T_ / 64), 256, 0, stream>>>(q_full, k_full, w_ws, scores);

  float* out_vals = (float*)d_out;
  int* out_idx = (int*)((float*)d_out + (size_t)T_ * TOPK_);
  topk_kernel<<<T_, 512, 0, stream>>>(scores, out_vals, out_idx);
}
